// Masked_Actor_Net_PNAConv_4999341932622
// MI455X (gfx1250) — matrix operand order repaired and checked by reading, not compiled, not run on hardware
//
#include <hip/hip_runtime.h>
#include <hip/hip_bf16.h>
#include <math.h>

// ---------------------------------------------------------------------------
// Masked_Actor_Net_PNAConv for MI455X (gfx1250, wave32, WMMA).
// bf16 WMMA (v_wmma_f32_16x16x32_bf16) tiled GEMM, f32 accumulate.
// Block tile 64(M) x 128(N): 8 waves, each wave owns one 16-col N tile and
// 4 M-subtiles (4 accumulators). LDS double-buffered; tiles stored pre-
// swizzled into WMMA fragment order so fragment loads are 2x ds_load_b128.
// All dims in this net: M % 64 == 0, Nc % 16 == 0, K % 32 == 0 -> no guards.
// ---------------------------------------------------------------------------

typedef __attribute__((ext_vector_type(16))) __bf16 v16bf;
typedef __attribute__((ext_vector_type(8)))  float  v8f;

#define PNA_N   16384   // B*NN
#define PNA_DEG 8
#define PNA_TP  192
#define PNA_B   64
#define PNA_NN  256

__device__ __forceinline__ float pna_act(float v, int act) {
    if (act == 1) return v > 0.f ? v : 0.f;                 // relu (== relu(leaky(x)))
    if (act == 2) return v > 0.f ? v : 0.01f * v;           // leaky_relu(0.01)
    if (act == 3) return 1.f / (1.f + __expf(-v));          // sigmoid
    return v;
}

// ----------------------------- WMMA GEMM -----------------------------------
// C[M,Nc] = act( bn(A)[M,K] @ B[K,Nc] + bias ), row-major, arbitrary ld's.
// bn(A) (optional): (A[:,k]-bnMean[k])*bnScale[k]+bnShift[k] applied on load.
// Requires M % 64 == 0, K % 32 == 0, Nc % 16 == 0 (true for this network).
__global__ void pna_gemm_bf16_wmma(const float* __restrict__ A,
                                   const float* __restrict__ Bm,
                                   const float* __restrict__ bias,
                                   const float* __restrict__ bnMean,
                                   const float* __restrict__ bnScale,
                                   const float* __restrict__ bnShift,
                                   float* __restrict__ C,
                                   int M, int Nc, int K,
                                   int lda, int ldb, int ldc, int act)
{
    // fragment-order (swizzled) tiles: element index = lane*16 + e
    __shared__ __align__(32) __bf16 sA[2][4 * 512];   // 4 M-subtiles of 16x32
    __shared__ __align__(32) __bf16 sB[2][8 * 512];   // 8 N-tiles of 32x16

    const int tid  = threadIdx.x;
    const int wave = tid >> 5;
    const int lane = tid & 31;
    const int half = lane >> 4;
    const int l16  = lane & 15;

    const int mBase  = blockIdx.y * 64;
    const int nTile0 = blockIdx.x * 8;
    const int nBase  = (nTile0 + wave) * 16;

    // A swizzle: element (m,k) of a 16x32 subtile is read by lane
    //   L = ((k&15)>>3)*16 + m   at fragment slot e = ((k>>4)*4 + ((k&7)>>1))*2 + (k&1)
    auto stageA = [&](int buf, int k0) {
#pragma unroll
        for (int it = 0; it < 2; ++it) {
            int i   = tid + it * 256;          // 512 float4 loads: 64 rows x 8
            int m64 = i >> 3;
            int k4  = (i & 7) << 2;            // 4 consecutive k
            float4 v = *(const float4*)&A[(size_t)(mBase + m64) * lda + k0 + k4];
            if (bnMean) {
                int c = k0 + k4;
                v.x = (v.x - bnMean[c    ]) * bnScale[c    ] + bnShift[c    ];
                v.y = (v.y - bnMean[c + 1]) * bnScale[c + 1] + bnShift[c + 1];
                v.z = (v.z - bnMean[c + 2]) * bnScale[c + 2] + bnShift[c + 2];
                v.w = (v.w - bnMean[c + 3]) * bnScale[c + 3] + bnShift[c + 3];
            }
            int sub = m64 >> 4, m = m64 & 15;
            int h  = (k4 >> 3) & 1;
            int e0 = (((k4 >> 4) << 2) + ((k4 & 7) >> 1)) << 1;
            __bf16* p = &sA[buf][sub * 512 + (((h << 4) | m) << 4) + e0];
            p[0] = (__bf16)v.x; p[1] = (__bf16)v.y;
            p[2] = (__bf16)v.z; p[3] = (__bf16)v.w;
        }
    };
    // B swizzle: element (k,n) of a 32x16 tile -> lane L = ((k>>3)&1)*16 + n, e = (k>>4)*8 + (k&7)
    auto stageB = [&](int buf, int k0) {
#pragma unroll
        for (int it = 0; it < 4; ++it) {
            int i  = tid + it * 256;           // 1024 float4 loads
            int t  = i >> 7, r = i & 127;
            int kk = r >> 2;
            int nn4 = (r & 3) << 2;
            int ng = (nTile0 + t) * 16 + nn4;
            if (ng > Nc - 4) ng = Nc - 4;      // dead-tile clamp (wave never stores)
            float4 v = *(const float4*)&Bm[(size_t)(k0 + kk) * ldb + ng];
            __bf16* p = &sB[buf][t * 512 + ((((kk >> 3) & 1) << 4) + nn4) * 16 + ((kk & 16) >> 1) + (kk & 7)];
            p[0]  = (__bf16)v.x; p[16] = (__bf16)v.y;
            p[32] = (__bf16)v.z; p[48] = (__bf16)v.w;
        }
    };

    v8f acc0 = {}, acc1 = {}, acc2 = {}, acc3 = {};
    stageA(0, 0); stageB(0, 0);
    int buf = 0;
    for (int k0 = 0; k0 < K; k0 += 32) {
        __syncthreads();
        if (k0 + 32 < K) { stageA(buf ^ 1, k0 + 32); stageB(buf ^ 1, k0 + 32); }
        v16bf b  = *(const v16bf*)&sB[buf][wave * 512 + lane * 16];
        v16bf a0 = *(const v16bf*)&sA[buf][          lane * 16];
        v16bf a1 = *(const v16bf*)&sA[buf][1 * 512 + lane * 16];
        v16bf a2 = *(const v16bf*)&sA[buf][2 * 512 + lane * 16];
        v16bf a3 = *(const v16bf*)&sA[buf][3 * 512 + lane * 16];
        acc0 = __builtin_amdgcn_wmma_f32_16x16x32_bf16(false, a0, false, b, (short)0, acc0, false, false);
        acc1 = __builtin_amdgcn_wmma_f32_16x16x32_bf16(false, a1, false, b, (short)0, acc1, false, false);
        acc2 = __builtin_amdgcn_wmma_f32_16x16x32_bf16(false, a2, false, b, (short)0, acc2, false, false);
        acc3 = __builtin_amdgcn_wmma_f32_16x16x32_bf16(false, a3, false, b, (short)0, acc3, false, false);
        buf ^= 1;
    }

    if (nBase < Nc) {                      // wave-uniform; live tiles are full
        const int n = nBase + l16;
        const float bv = bias ? bias[n] : 0.f;
#pragma unroll
        for (int s = 0; s < 4; ++s) {
            v8f acc = (s == 0) ? acc0 : (s == 1) ? acc1 : (s == 2) ? acc2 : acc3;
            int m0 = mBase + s * 16 + (half << 3);
#pragma unroll
            for (int r = 0; r < 8; ++r)
                C[(size_t)(m0 + r) * ldc + n] = pna_act(acc[r] + bv, act);
        }
    }
}

// ------------------------- PNA aggregation ----------------------------------
// For node n (one block): msg_k[c] = HS[src_k][c] + HD[n][c] + bm[c]
//                                    + sum_j es[8n+k][j] * WmE[j][c]
// Writes U[n] = [ h | mean | max | sum | std ]  (5*cin columns).
__global__ void pna_aggregate(const float* __restrict__ h,
                              const float* __restrict__ HS,
                              const float* __restrict__ HD,
                              const float* __restrict__ es,
                              const float* __restrict__ WmE,
                              const float* __restrict__ bm,
                              const int*   __restrict__ src,
                              float* __restrict__ U, int cin)
{
    const int n = blockIdx.x;
    __shared__ float sE[PNA_DEG * 16];
    __shared__ int   sSrc[PNA_DEG];
    const int tid = threadIdx.x;
    if (tid < PNA_DEG * 16) sE[tid] = es[(size_t)n * PNA_DEG * 16 + tid];
    if (tid < PNA_DEG)      sSrc[tid] = src[n * PNA_DEG + tid];
    __syncthreads();

    const size_t base = (size_t)n * cin;
    const size_t ub   = (size_t)n * 5 * cin;
    for (int c = tid; c < cin; c += blockDim.x) {
        float we[16];
#pragma unroll
        for (int j = 0; j < 16; ++j) we[j] = WmE[(size_t)j * cin + c];
        const float hd = HD[base + c] + bm[c];
        float s = 0.f, ssq = 0.f, mx = -3.402823466e38f;
#pragma unroll
        for (int k = 0; k < PNA_DEG; ++k) {
            float v = HS[(size_t)sSrc[k] * cin + c] + hd;
#pragma unroll
            for (int j = 0; j < 16; ++j) v += sE[k * 16 + j] * we[j];
            s += v; ssq += v * v; mx = fmaxf(mx, v);
        }
        const float mean = s * 0.125f;
        const float var  = ssq * 0.125f - mean * mean;
        const float stdv = sqrtf(fmaxf(var, 0.f) + 1e-30f);
        U[ub + c]           = h[base + c];
        U[ub + cin + c]     = mean;
        U[ub + 2 * cin + c] = mx;
        U[ub + 3 * cin + c] = s;
        U[ub + 4 * cin + c] = stdv;
    }
}

// --------------------------- BatchNorm stats --------------------------------
__global__ void pna_bn_stats(const float* __restrict__ u,
                             const float* __restrict__ bng,
                             float* __restrict__ scale,
                             float* __restrict__ meanv,
                             int n, int cout)
{
    const int c = blockIdx.x;
    __shared__ float ss[256], sq[256];
    float s = 0.f, q = 0.f;
    for (int r = threadIdx.x; r < n; r += 256) {
        float v = u[(size_t)r * cout + c];
        s += v; q += v * v;
    }
    ss[threadIdx.x] = s; sq[threadIdx.x] = q;
    __syncthreads();
    for (int o = 128; o > 0; o >>= 1) {
        if (threadIdx.x < o) { ss[threadIdx.x] += ss[threadIdx.x + o];
                               sq[threadIdx.x] += sq[threadIdx.x + o]; }
        __syncthreads();
    }
    if (threadIdx.x == 0) {
        float mean = ss[0] / (float)n;
        float var  = sq[0] / (float)n - mean * mean;
        scale[c] = bng[c] * rsqrtf(var + 1e-5f);
        meanv[c] = mean;
    }
}

// ------------------------------ head kernels --------------------------------
__global__ void pna_node_max(const float* __restrict__ h3, float* __restrict__ nm, int n)
{
    int i = blockIdx.x * blockDim.x + threadIdx.x;
    if (i < n) {
        float m = -3.402823466e38f;
        const float* row = h3 + (size_t)i * PNA_TP;
        for (int t = 0; t < PNA_TP; ++t) m = fmaxf(m, row[t]);
        nm[i] = m;
    }
}

__global__ void pna_feat_mask(const float* __restrict__ g,
                              const float* __restrict__ h3,
                              const float* __restrict__ mask,
                              float* __restrict__ out, size_t total)
{
    size_t i = (size_t)blockIdx.x * blockDim.x + threadIdx.x;
    if (i < total) {
        size_t node = i / PNA_TP;
        float v = g[node] * h3[i];
        if (mask[i] == 0.f) v = -1e5f;
        out[i] = v;
    }
}

__global__ void pna_softmax_row(float* __restrict__ out, int cols)
{
    float* row = out + (size_t)blockIdx.x * cols;
    __shared__ float red[256];
    float mx = -3.402823466e38f;
    for (int i = threadIdx.x; i < cols; i += 256) mx = fmaxf(mx, row[i]);
    red[threadIdx.x] = mx; __syncthreads();
    for (int o = 128; o > 0; o >>= 1) {
        if (threadIdx.x < o) red[threadIdx.x] = fmaxf(red[threadIdx.x], red[threadIdx.x + o]);
        __syncthreads();
    }
    mx = red[0]; __syncthreads();
    float s = 0.f;
    for (int i = threadIdx.x; i < cols; i += 256) s += __expf(row[i] - mx);
    red[threadIdx.x] = s; __syncthreads();
    for (int o = 128; o > 0; o >>= 1) {
        if (threadIdx.x < o) red[threadIdx.x] += red[threadIdx.x + o];
        __syncthreads();
    }
    float inv = 1.f / red[0];
    for (int i = threadIdx.x; i < cols; i += 256) row[i] = __expf(row[i] - mx) * inv;
}

// ------------------------------- orchestration ------------------------------
extern "C" void kernel_launch(void* const* d_in, const int* in_sizes, int n_in,
                              void* d_out, int out_size, void* d_ws, size_t ws_size,
                              hipStream_t stream)
{
    (void)in_sizes; (void)n_in; (void)out_size; (void)ws_size;
    const float* ns      = (const float*)d_in[0];
    const float* es      = (const float*)d_in[1];
    const float* dm      = (const float*)d_in[2];
    const float* mask_fv = (const float*)d_in[3];
    const int*   src     = (const int*)  d_in[4];
    // d_in[5] = dst (implied: node n owns edges 8n..8n+7)
    const float* Wm[3] = {(const float*)d_in[6],  (const float*)d_in[14], (const float*)d_in[22]};
    const float* bm[3] = {(const float*)d_in[7],  (const float*)d_in[15], (const float*)d_in[23]};
    const float* Wu[3] = {(const float*)d_in[8],  (const float*)d_in[16], (const float*)d_in[24]};
    const float* bu[3] = {(const float*)d_in[9],  (const float*)d_in[17], (const float*)d_in[25]};
    const float* bng[3]= {(const float*)d_in[10], (const float*)d_in[18], (const float*)d_in[26]};
    const float* bnb[3]= {(const float*)d_in[11], (const float*)d_in[19], (const float*)d_in[27]};
    const float* Wx[3] = {(const float*)d_in[12], (const float*)d_in[20], (const float*)d_in[28]};
    const float* bx[3] = {(const float*)d_in[13], (const float*)d_in[21], (const float*)d_in[29]};
    const float* W1 = (const float*)d_in[30]; const float* b1 = (const float*)d_in[31];
    const float* W2 = (const float*)d_in[32]; const float* b2 = (const float*)d_in[33];
    const float* W3 = (const float*)d_in[34]; const float* b3 = (const float*)d_in[35];
    const float* W4 = (const float*)d_in[36]; const float* b4 = (const float*)d_in[37];
    float* out = (float*)d_out;

    const size_t Nn = PNA_N;
    // workspace arena (floats); peak ~64M floats (~256 MB)
    float* ws = (float*)d_ws;
    size_t off = 0;
    auto alloc = [&](size_t n) { float* p = ws + off; off += n; return p; };
    float* h2in  = alloc(Nn * 416);   // [h1 | d2]
    float* h2    = alloc(Nn * 384);
    float* h3    = alloc(Nn * 192);
    float* HS    = alloc(Nn * 416);
    float* HD    = alloc(Nn * 416);
    float* U     = alloc(Nn * 2080);
    float* scale = alloc(512);
    float* meanv = alloc(512);
    float* nm    = alloc(Nn);
    float* t2    = alloc(64 * 64);
    float* g     = alloc(64 * 256);
    float* u  = HS;   // alias: HS dead once aggregation finished
    float* t1 = HD;   // alias: HD dead between layers

    auto gemm = [&](const float* A, const float* Bmat, const float* bias,
                    const float* bnM, const float* bnS, const float* bnSh,
                    float* C, int M, int Nc, int K,
                    int lda, int ldb, int ldc, int act) {
        dim3 grid((Nc / 16 + 7) / 8, M / 64);
        pna_gemm_bf16_wmma<<<grid, 256, 0, stream>>>(A, Bmat, bias, bnM, bnS, bnSh,
                                                     C, M, Nc, K, lda, ldb, ldc, act);
    };

    const int cins[3]  = {128, 416, 384};
    const int couts[3] = {384, 384, 192};
    const float* hin[3] = {ns, h2in, h2};
    const int   hld[3]  = {128, 416, 384};
    float* hout[3]      = {h2in /*ldc 416*/, h2, h3};
    const int outld[3]  = {416, 384, 192};
    const int outact[3] = {1, 1, 2};   // relu, relu, leaky

    for (int L = 0; L < 3; ++L) {
        const int cin = cins[L], cout = couts[L];
        // HS = h @ Wm[:cin], HD = h @ Wm[cin:2cin]
        gemm(hin[L], Wm[L],                     nullptr, nullptr, nullptr, nullptr,
             HS, PNA_N, cin, cin, hld[L], cin, cin, 0);
        gemm(hin[L], Wm[L] + (size_t)cin * cin, nullptr, nullptr, nullptr, nullptr,
             HD, PNA_N, cin, cin, hld[L], cin, cin, 0);
        // aggregate (fuses e @ Wm[2cin:] + bm)
        pna_aggregate<<<PNA_N, 256, 0, stream>>>(hin[L], HS, HD, es,
                                                 Wm[L] + (size_t)2 * cin * cin, bm[L],
                                                 src, U, cin);
        // u = U @ Wu + bu
        gemm(U, Wu[L], bu[L], nullptr, nullptr, nullptr,
             u, PNA_N, cout, 5 * cin, 5 * cin, cout, cout, 0);
        // BatchNorm stats (training: over all N rows)
        pna_bn_stats<<<cout, 256, 0, stream>>>(u, bng[L], scale, meanv, PNA_N, cout);
        // h_next = act( bn(u) @ Wx + bx )   (BN applied in GEMM A-load)
        gemm(u, Wx[L], bx[L], meanv, scale, bnb[L],
             hout[L], PNA_N, cout, cout, cout, cout, outld[L], outact[L]);

        if (L == 0) {
            // d2 = (dm @ W1 + b1) @ W2 + b2  -> written into h2in[:, 384:416]
            gemm(dm, W1, b1, nullptr, nullptr, nullptr, t1, PNA_N, 64, 256, 256, 64, 64, 0);
            gemm(t1, W2, b2, nullptr, nullptr, nullptr, h2in + 384, PNA_N, 32, 64, 64, 32, 416, 0);
        }
    }

    // head: nm = h3.max(-1); g = sigmoid(relu(nm@W3+b3)@W4+b4)
    pna_node_max<<<(PNA_N + 255) / 256, 256, 0, stream>>>(h3, nm, PNA_N);
    gemm(nm, W3, b3, nullptr, nullptr, nullptr, t2, PNA_B, 64, PNA_NN, PNA_NN, 64, 64, 1);
    gemm(t2, W4, b4, nullptr, nullptr, nullptr, g,  PNA_B, PNA_NN, 64, 64, PNA_NN, PNA_NN, 3);
    // feat = g[node] * h3, mask, softmax over NN*TP per batch row
    {
        size_t tot = (size_t)PNA_N * PNA_TP;
        pna_feat_mask<<<(unsigned)((tot + 255) / 256), 256, 0, stream>>>(g, h3, mask_fv, out, tot);
    }
    pna_softmax_row<<<PNA_B, 256, 0, stream>>>(out, PNA_NN * PNA_TP);
}
